// GestureSmoothingLSTM_3315714752883
// MI455X (gfx1250) — matrix operand order repaired and checked by reading, not compiled, not run on hardware
//
#include <hip/hip_runtime.h>

// ---------------------------------------------------------------------------
// GestureSmoothingLSTM on MI455X (gfx1250), wave32 + WMMA bf16.
//
// Pipeline (all on `stream`, in-order):
//   1) pack_b_kernel x4   : f32 weights -> bf16 WMMA B-tile layout in d_ws
//   2) input_proj_kernel  : x0 = concat(emb[s],emb[e],sc,ec) @ W_ih0^T + b   (f32, in d_ws)
//   3) lstm_seq_kernel    : 100-step 2-layer LSTM recurrence + fc, WMMA bf16
//
// d_ws layout (bytes):
//   [0, 32MiB)          x0        : f32 [4096][2048]
//   [32MiB, +2MiB)      Whh0 pack : bf16 tiles
//   [.. +2MiB)          Wih1 pack
//   [.. +2MiB)          Whh1 pack
//   [.. +64KiB)         fcW  pack (N padded 63->64)
// ---------------------------------------------------------------------------

#define HID   512
#define G4    2048          // 4*HID gate columns
#define ROWS  16            // batch rows per workgroup
#define SEQT  100

typedef __attribute__((ext_vector_type(16))) __bf16 v16bf;
typedef __attribute__((ext_vector_type(8)))  __bf16 v8bf;
typedef __attribute__((ext_vector_type(8)))  float  v8f;

static __device__ __forceinline__ v8f wmma_bf16(v16bf a, v16bf b, v8f c) {
  // (neg_a, A, neg_b, B, c_mod, C, reuse_a, reuse_b)
  return __builtin_amdgcn_wmma_f32_16x16x32_bf16(false, a, false, b, (short)0, c,
                                                 false, false);
}

// A-operand (16x32 bf16, lane = M row) built from row-major bf16 LDS buffer.
// Per ISA 16-bit A layout: lane group 0: K = {k..k+7, k+16..k+23};
//                          lane group 1: K = {k+8..k+15, k+24..k+31}.
// Caller passes row pointer (base + M*HID) and k0 = kt*32 + 8*grp.
static __device__ __forceinline__ v16bf ldsA(const __bf16* __restrict__ row, int k0) {
  v8bf lo = *(const v8bf*)(row + k0);
  v8bf hi = *(const v8bf*)(row + k0 + 16);
  return __builtin_shufflevector(lo, hi, 0,1,2,3,4,5,6,7,8,9,10,11,12,13,14,15);
}

// Packed B-operand: one 32x16 tile = 1KB, lane's 16 bf16 are contiguous.
static __device__ __forceinline__ v16bf gB(const __bf16* __restrict__ p) {
  return *(const v16bf*)p;
}

static __device__ __forceinline__ float sigmf(float x) {
  return __builtin_amdgcn_rcpf(1.0f + __expf(-x));
}
static __device__ __forceinline__ float tanh_f(float x) {
  float e = __expf(-2.0f * x);
  return (1.0f - e) * __builtin_amdgcn_rcpf(1.0f + e);
}

// ---------------------------------------------------------------------------
// Pack f32 weight [realN x realK] row-major into bf16 WMMA B tiles.
// Tile (nt,kt): B[k,n] = W[n,k]; lane l: N = nt*16 + (l&15), K = kt*32 + 8*(l>>4) + (e<8 ? e : e+8).
// Tile storage: out[(nt*KT + kt)*512 + l*16 + e].
// ---------------------------------------------------------------------------
__global__ __launch_bounds__(256)
void pack_b_kernel(const float* __restrict__ W, __bf16* __restrict__ out,
                   int N, int K, int realN, int realK) {
  int tid = blockIdx.x * blockDim.x + threadIdx.x;
  int KT = K >> 5;
  int NT = N >> 4;
  if (tid >= NT * KT * 32) return;
  int lane = tid & 31;
  int tile = tid >> 5;
  int kt = tile % KT;
  int nt = tile / KT;
  int n  = nt * 16 + (lane & 15);
  int k0 = kt * 32 + 8 * (lane >> 4);
  __bf16 vals[16];
#pragma unroll
  for (int e = 0; e < 16; ++e) {
    int k = k0 + (e < 8 ? e : e + 8);
    float f = (n < realN && k < realK) ? W[(size_t)n * realK + k] : 0.0f;
    vals[e] = (__bf16)f;
  }
  __bf16* dst = out + (size_t)tile * 512 + lane * 16;
#pragma unroll
  for (int e = 0; e < 16; ++e) dst[e] = vals[e];
}

// ---------------------------------------------------------------------------
// x0[b][n] = dot(x[b,:254], W_ih0[n,:254]) + b_ih0[n] + b_hh0[n]
// x[b] = concat(emb[sl[b]], emb[el[b]], sc[b], ec[b])
// Block: 16 rows x 256 cols. grid = (4096/16, 2048/256).
// ---------------------------------------------------------------------------
__global__ __launch_bounds__(256)
void input_proj_kernel(const int* __restrict__ sl, const int* __restrict__ el,
                       const float* __restrict__ sc, const float* __restrict__ ec,
                       const float* __restrict__ emb, const float* __restrict__ Wih0,
                       const float* __restrict__ bih0, const float* __restrict__ bhh0,
                       float* __restrict__ x0) {
  __shared__ float xs[254][17];   // [input_idx][row], padded to kill bank conflicts
  int rb = blockIdx.x * 16;
  int cb = blockIdx.y * 256;

  for (int idx = threadIdx.x; idx < 16 * 254; idx += 256) {
    int r = idx / 254;
    int c = idx % 254;
    int b = rb + r;
    float v;
    if (c < 64)       v = emb[sl[b] * 64 + c];
    else if (c < 128) v = emb[el[b] * 64 + (c - 64)];
    else if (c < 191) v = sc[b * 63 + (c - 128)];
    else              v = ec[b * 63 + (c - 191)];
    xs[c][r] = v;
  }
  __syncthreads();

  int n = cb + threadIdx.x;
  float bias = bih0[n] + bhh0[n];
  float acc[16];
#pragma unroll
  for (int r = 0; r < 16; ++r) acc[r] = bias;

  const float* wr = Wih0 + (size_t)n * 254;
  for (int i = 0; i < 254; ++i) {
    float w = wr[i];
#pragma unroll
    for (int r = 0; r < 16; ++r) acc[r] += w * xs[i][r];
  }
  float* o = x0 + (size_t)rb * G4 + n;
#pragma unroll
  for (int r = 0; r < 16; ++r) o[(size_t)r * G4] = acc[r];
}

// ---------------------------------------------------------------------------
// Persistent 2-layer LSTM recurrence. One WG = 16 batch rows, 8 waves split
// the 2048 gate columns (wave cg owns hidden units [cg*64, cg*64+64), all 4
// gates). h0/h1 double-buffered in LDS (bf16); c0/c1 stay in VGPRs.
// __launch_bounds__(256, 3): force the VGPR allocation under the 256-reg MSB
// boundary (file/3) to kill s_set_vgpr_msb churn and raise occupancy; spills
// would appear as scratch ops in the histogram.
// ---------------------------------------------------------------------------
__global__ __launch_bounds__(256, 3)
void lstm_seq_kernel(const float* __restrict__ x0,
                     const __bf16* __restrict__ Wh0,
                     const __bf16* __restrict__ Wi1,
                     const __bf16* __restrict__ Wh1,
                     const __bf16* __restrict__ Wfc,
                     const float* __restrict__ bih1,
                     const float* __restrict__ bhh1,
                     const float* __restrict__ fcb,
                     float* __restrict__ out) {
  __shared__ __align__(16) __bf16 h0b[2][ROWS * HID];
  __shared__ __align__(16) __bf16 h1b[2][ROWS * HID];

  const int tid  = threadIdx.x;
  const int lane = tid & 31;
  const int ln   = lane & 15;   // N index within tile / A row index
  const int grp  = lane >> 4;   // half-wave group
  const int cg   = tid >> 5;    // wave id 0..7 = column group
  const int rowbase = blockIdx.x * ROWS;

  // zero initial (p=0) h buffers
  {
    unsigned int* z0 = (unsigned int*)&h0b[0][0];
    unsigned int* z1 = (unsigned int*)&h1b[0][0];
    for (int i = tid; i < ROWS * HID / 2; i += 256) { z0[i] = 0u; z1[i] = 0u; }
  }

  v8f c0[4] = {};   // cell state, layer 0 (4 hidden tiles, D layout)
  v8f c1[4] = {};   // cell state, layer 1

  // layer-1 bias (b_ih1 + b_hh1), per lane per hidtile per gate
  float b1v[4][4];
#pragma unroll
  for (int h = 0; h < 4; ++h) {
    int c = cg * 64 + h * 16 + ln;
#pragma unroll
    for (int g = 0; g < 4; ++g) b1v[h][g] = bih1[c + g * 512] + bhh1[c + g * 512];
  }
  const int   fcn    = (cg & 3) * 16 + ln;
  const float fcbias = (fcn < 63) ? fcb[fcn] : 0.0f;

  const float* x0base = x0 + (size_t)(rowbase + 8 * grp) * G4 + ln;

  __syncthreads();

  int p = 0;
  for (int t = 0; t < SEQT; ++t) {
    const __bf16* Ah0 = &h0b[p][ln * HID];       // old h0, A rows
    const __bf16* Ah1 = &h1b[p][ln * HID];       // old h1, A rows
    __bf16* Wr0 = &h0b[p ^ 1][0];                // new h0 buffer
    __bf16* Wr1 = &h1b[p ^ 1][0];                // new h1 buffer

    // ---------------- layer 0: pre = x0 + h0 @ Whh0^T ----------------
#pragma unroll
    for (int h = 0; h < 4; ++h) {
      const int col0 = cg * 64 + h * 16;
      v8f aI, aF, aG, aO;
#pragma unroll
      for (int v = 0; v < 8; ++v) {
        const float* r = x0base + (size_t)v * G4;
        aI[v] = r[col0];
        aF[v] = r[col0 + 512];
        aG[v] = r[col0 + 1024];
        aO[v] = r[col0 + 1536];
      }
      const __bf16* wI = Wh0 + ((size_t)(col0 >> 4) * 16) * 512 + lane * 16;
#pragma unroll 2
      for (int kt = 0; kt < 16; ++kt) {
        v16bf A = ldsA(Ah0, kt * 32 + 8 * grp);
        aI = wmma_bf16(A, gB(wI + (size_t)kt * 512),             aI);
        aF = wmma_bf16(A, gB(wI + (size_t)(32 * 16 + kt) * 512), aF);
        aG = wmma_bf16(A, gB(wI + (size_t)(64 * 16 + kt) * 512), aG);
        aO = wmma_bf16(A, gB(wI + (size_t)(96 * 16 + kt) * 512), aO);
      }
      __bf16* hw = Wr0 + col0 + ln;
#pragma unroll
      for (int v = 0; v < 8; ++v) {
        float ii = sigmf(aI[v]);
        float ff = sigmf(aF[v]);
        float gg = tanh_f(aG[v]);
        float oo = sigmf(aO[v]);
        float cn = ff * c0[h][v] + ii * gg;
        c0[h][v] = cn;
        hw[(v + 8 * grp) * HID] = (__bf16)(oo * tanh_f(cn));
      }
    }
    __syncthreads();   // new h0 visible

    // ------- layer 1: pre = h0new @ Wih1^T + b1 + h1old @ Whh1^T -------
    const __bf16* Ah0n = &h0b[p ^ 1][ln * HID];
#pragma unroll
    for (int h = 0; h < 4; ++h) {
      const int col0 = cg * 64 + h * 16;
      v8f aI, aF, aG, aO;
#pragma unroll
      for (int v = 0; v < 8; ++v) {
        aI[v] = b1v[h][0]; aF[v] = b1v[h][1]; aG[v] = b1v[h][2]; aO[v] = b1v[h][3];
      }
      const __bf16* wA = Wi1 + ((size_t)(col0 >> 4) * 16) * 512 + lane * 16;
      const __bf16* wB = Wh1 + ((size_t)(col0 >> 4) * 16) * 512 + lane * 16;
#pragma unroll 2
      for (int kt = 0; kt < 16; ++kt) {
        v16bf A = ldsA(Ah0n, kt * 32 + 8 * grp);
        aI = wmma_bf16(A, gB(wA + (size_t)kt * 512),             aI);
        aF = wmma_bf16(A, gB(wA + (size_t)(32 * 16 + kt) * 512), aF);
        aG = wmma_bf16(A, gB(wA + (size_t)(64 * 16 + kt) * 512), aG);
        aO = wmma_bf16(A, gB(wA + (size_t)(96 * 16 + kt) * 512), aO);
      }
#pragma unroll 2
      for (int kt = 0; kt < 16; ++kt) {
        v16bf A = ldsA(Ah1, kt * 32 + 8 * grp);
        aI = wmma_bf16(A, gB(wB + (size_t)kt * 512),             aI);
        aF = wmma_bf16(A, gB(wB + (size_t)(32 * 16 + kt) * 512), aF);
        aG = wmma_bf16(A, gB(wB + (size_t)(64 * 16 + kt) * 512), aG);
        aO = wmma_bf16(A, gB(wB + (size_t)(96 * 16 + kt) * 512), aO);
      }
      __bf16* hw = Wr1 + col0 + ln;
#pragma unroll
      for (int v = 0; v < 8; ++v) {
        float ii = sigmf(aI[v]);
        float ff = sigmf(aF[v]);
        float gg = tanh_f(aG[v]);
        float oo = sigmf(aO[v]);
        float cn = ff * c1[h][v] + ii * gg;
        c1[h][v] = cn;
        hw[(v + 8 * grp) * HID] = (__bf16)(oo * tanh_f(cn));
      }
    }
    __syncthreads();   // new h1 visible

    // ---------------- fc: y = h1new @ fcW^T + fcb ----------------
    if (cg < 4) {
      const __bf16* Ah1n = &h1b[p ^ 1][ln * HID];
      v8f acc;
#pragma unroll
      for (int v = 0; v < 8; ++v) acc[v] = fcbias;
      const __bf16* wF = Wfc + ((size_t)cg * 16) * 512 + lane * 16;
#pragma unroll 2
      for (int kt = 0; kt < 16; ++kt)
        acc = wmma_bf16(ldsA(Ah1n, kt * 32 + 8 * grp),
                        gB(wF + (size_t)kt * 512), acc);
      if (fcn < 63) {
#pragma unroll
        for (int v = 0; v < 8; ++v)
          out[(size_t)(rowbase + v + 8 * grp) * 6300 + fcn * 100 + t] = acc[v];
      }
    }
    p ^= 1;
  }
}

// ---------------------------------------------------------------------------
extern "C" void kernel_launch(void* const* d_in, const int* in_sizes, int n_in,
                              void* d_out, int out_size, void* d_ws, size_t ws_size,
                              hipStream_t stream) {
  (void)in_sizes; (void)n_in; (void)out_size; (void)ws_size;
  const int*   sl   = (const int*)d_in[0];
  const int*   el   = (const int*)d_in[1];
  const float* sc   = (const float*)d_in[2];
  const float* ec   = (const float*)d_in[3];
  const float* emb  = (const float*)d_in[4];
  const float* Wih0 = (const float*)d_in[5];
  const float* Whh0 = (const float*)d_in[6];
  const float* bih0 = (const float*)d_in[7];
  const float* bhh0 = (const float*)d_in[8];
  const float* Wih1 = (const float*)d_in[9];
  const float* Whh1 = (const float*)d_in[10];
  const float* bih1 = (const float*)d_in[11];
  const float* bhh1 = (const float*)d_in[12];
  const float* fcW  = (const float*)d_in[13];
  const float* fcb  = (const float*)d_in[14];
  float* out = (float*)d_out;

  char*   ws   = (char*)d_ws;
  float*  x0   = (float*)ws;                                   // 32 MiB
  __bf16* Wh0p = (__bf16*)(ws + 33554432);                     // 2 MiB
  __bf16* Wi1p = (__bf16*)(ws + 33554432 + 1 * 2097152);       // 2 MiB
  __bf16* Wh1p = (__bf16*)(ws + 33554432 + 2 * 2097152);       // 2 MiB
  __bf16* Wfcp = (__bf16*)(ws + 33554432 + 3 * 2097152);       // 64 KiB

  pack_b_kernel<<<256, 256, 0, stream>>>(Whh0, Wh0p, 2048, 512, 2048, 512);
  pack_b_kernel<<<256, 256, 0, stream>>>(Wih1, Wi1p, 2048, 512, 2048, 512);
  pack_b_kernel<<<256, 256, 0, stream>>>(Whh1, Wh1p, 2048, 512, 2048, 512);
  pack_b_kernel<<<8,   256, 0, stream>>>(fcW,  Wfcp,   64, 512,   63, 512);
  input_proj_kernel<<<dim3(256, 8), 256, 0, stream>>>(sl, el, sc, ec, emb, Wih0,
                                                      bih0, bhh0, x0);
  lstm_seq_kernel<<<256, 256, 0, stream>>>(x0, Wh0p, Wi1p, Wh1p, Wfcp,
                                           bih1, bhh1, fcb, out);
}
